// NAR_26749056319699
// MI455X (gfx1250) — matrix operand order repaired and checked by reading, not compiled, not run on hardware
//
#include <hip/hip_runtime.h>

// ---------------------------------------------------------------------------
// CDNA5 (gfx1250) GCN forward: software-pipelined WMMA f16->f32 GEMMs
// + L2-resident fp32 atomic scatter for the edge aggregation.
// ---------------------------------------------------------------------------

typedef __attribute__((ext_vector_type(16))) _Float16 v16h;
typedef __attribute__((ext_vector_type(8)))  _Float16 v8h;
typedef __attribute__((ext_vector_type(8)))  float    v8f;

static constexpr int NN    = 10000;   // nodes
static constexpr int EE    = 320000;  // edges
static constexpr int DD    = 256;     // embed dim
static constexpr int HH    = 512;     // hidden dim
static constexpr int MPAD  = 10016;   // NN rounded up to 32 (wave computes 32 rows)

// ---------------- WMMA fragment loaders (wave32, contiguous k = 16h + s in A and B) --------
// A (row-major M x K, f16): lane m in [0,15] -> row m0+m; elems j = K[kh+j],
// where kh = (lane>=16) ? 16 : 0.
__device__ __forceinline__ v16h load_a_frag(const _Float16* __restrict__ A,
                                            int lda, int m0, int k0, int lane) {
    const int m  = m0 + (lane & 15);
    const int kh = (lane >> 4) << 4;
    const _Float16* p = A + (size_t)m * lda + k0 + kh;
    union { v16h v; v8h h[2]; } u;
    u.h[0] = *(const v8h*)(p);
    u.h[1] = *(const v8h*)(p + 8);
    return u.v;
}

// B stored transposed as Bt[N][K] (f16): lane n in [0,15] -> col n0+n;
// elems j = K[k0 + kh + j], kh = (lane>=16) ? 16 : 0 -> one 32B contiguous load.
__device__ __forceinline__ v16h load_b_frag(const _Float16* __restrict__ Bt,
                                            int ldb, int n0, int k0, int lane) {
    const int n  = n0 + (lane & 15);
    const int kh = (lane >> 4) << 4;
    return *(const v16h*)(Bt + (size_t)n * ldb + k0 + kh);
}

__device__ __forceinline__ v8f wmma_f16(v16h a, v16h b, v8f c) {
    return __builtin_amdgcn_wmma_f32_16x16x32_f16(false, a, false, b, (short)0, c, false, false);
}

// ---------------- GEMM: wave -> 32x64 strip, K templated, SW-pipelined -----
enum { EPI_SCALE_ROW_F32 = 0, EPI_BIAS_RELU_F16 = 1, EPI_BIAS_F32 = 2 };

template <int EPI, int K>
__global__ __launch_bounds__(256)
void gemm_wmma(const _Float16* __restrict__ A, const _Float16* __restrict__ Bt,
               const float* __restrict__ aux,   // dinv[M] (EPI 0) or bias[Ncols] (EPI 1,2)
               void* __restrict__ Cout, int M, int Ncols, int Mstore)
{
    const int lane   = threadIdx.x & 31;
    const int wave   = blockIdx.x * (blockDim.x >> 5) + (threadIdx.x >> 5);
    const int mtiles = M >> 5;                 // 32-row tiles
    const int ngrp   = Ncols >> 6;             // 64-wide N groups
    if (wave >= mtiles * ngrp) return;
    // consecutive waves (same block) -> adjacent M tiles, same N group (B reuse)
    const int ng = wave / mtiles;
    const int mt = wave - ng * mtiles;
    const int m0 = mt << 5;
    const int n0 = ng << 6;

    v8f c00 = {}, c01 = {}, c02 = {}, c03 = {};
    v8f c10 = {}, c11 = {}, c12 = {}, c13 = {};

    // prologue: fragments for k0 = 0
    v16h a0 = load_a_frag(A, K, m0,      0, lane);
    v16h a1 = load_a_frag(A, K, m0 + 16, 0, lane);
    v16h b0 = load_b_frag(Bt, K, n0 +  0, 0, lane);
    v16h b1 = load_b_frag(Bt, K, n0 + 16, 0, lane);
    v16h b2 = load_b_frag(Bt, K, n0 + 32, 0, lane);
    v16h b3 = load_b_frag(Bt, K, n0 + 48, 0, lane);

#pragma unroll
    for (int k0 = 0; k0 < K; k0 += 32) {
        const int kn = k0 + 32;
        v16h na0, na1, nb0, nb1, nb2, nb3;
        if (kn < K) {   // issue next-iteration loads before consuming current frags
            na0 = load_a_frag(A, K, m0,      kn, lane);
            na1 = load_a_frag(A, K, m0 + 16, kn, lane);
            nb0 = load_b_frag(Bt, K, n0 +  0, kn, lane);
            nb1 = load_b_frag(Bt, K, n0 + 16, kn, lane);
            nb2 = load_b_frag(Bt, K, n0 + 32, kn, lane);
            nb3 = load_b_frag(Bt, K, n0 + 48, kn, lane);
        }
        c00 = wmma_f16(a0, b0, c00);
        c01 = wmma_f16(a0, b1, c01);
        c02 = wmma_f16(a0, b2, c02);
        c03 = wmma_f16(a0, b3, c03);
        c10 = wmma_f16(a1, b0, c10);
        c11 = wmma_f16(a1, b1, c11);
        c12 = wmma_f16(a1, b2, c12);
        c13 = wmma_f16(a1, b3, c13);
        if (kn < K) {
            a0 = na0; a1 = na1;
            b0 = nb0; b1 = nb1; b2 = nb2; b3 = nb3;
        }
    }

    // C/D layout: lanes 0-15 vgpr r -> row +r; lanes 16-31 vgpr r -> row +8+r
    const int col   = n0 + (lane & 15);
    const int rb0   = m0 + ((lane >> 4) << 3);        // rows for c0x
    const int rb1   = rb0 + 16;                       // rows for c1x

    if (EPI == EPI_SCALE_ROW_F32) {
        float* C = (float*)Cout;
#pragma unroll
        for (int r = 0; r < 8; ++r) {
            {
                const float  s    = aux[rb0 + r];
                const size_t base = (size_t)(rb0 + r) * Ncols;
                C[base + col +  0] = c00[r] * s;
                C[base + col + 16] = c01[r] * s;
                C[base + col + 32] = c02[r] * s;
                C[base + col + 48] = c03[r] * s;
            }
            {
                const float  s    = aux[rb1 + r];
                const size_t base = (size_t)(rb1 + r) * Ncols;
                C[base + col +  0] = c10[r] * s;
                C[base + col + 16] = c11[r] * s;
                C[base + col + 32] = c12[r] * s;
                C[base + col + 48] = c13[r] * s;
            }
        }
    } else if (EPI == EPI_BIAS_RELU_F16) {
        _Float16* C = (_Float16*)Cout;
        const float bv0 = aux[col +  0], bv1 = aux[col + 16];
        const float bv2 = aux[col + 32], bv3 = aux[col + 48];
#pragma unroll
        for (int r = 0; r < 8; ++r) {
            const size_t base0 = (size_t)(rb0 + r) * Ncols;
            const size_t base1 = (size_t)(rb1 + r) * Ncols;
            C[base0 + col +  0] = (_Float16)fmaxf(c00[r] + bv0, 0.0f);
            C[base0 + col + 16] = (_Float16)fmaxf(c01[r] + bv1, 0.0f);
            C[base0 + col + 32] = (_Float16)fmaxf(c02[r] + bv2, 0.0f);
            C[base0 + col + 48] = (_Float16)fmaxf(c03[r] + bv3, 0.0f);
            C[base1 + col +  0] = (_Float16)fmaxf(c10[r] + bv0, 0.0f);
            C[base1 + col + 16] = (_Float16)fmaxf(c11[r] + bv1, 0.0f);
            C[base1 + col + 32] = (_Float16)fmaxf(c12[r] + bv2, 0.0f);
            C[base1 + col + 48] = (_Float16)fmaxf(c13[r] + bv3, 0.0f);
        }
    } else {  // EPI_BIAS_F32 (row-guarded store: final output)
        float* C = (float*)Cout;
        const float bv0 = aux[col +  0], bv1 = aux[col + 16];
        const float bv2 = aux[col + 32], bv3 = aux[col + 48];
#pragma unroll
        for (int r = 0; r < 8; ++r) {
            if (rb0 + r < Mstore) {
                const size_t base = (size_t)(rb0 + r) * Ncols;
                C[base + col +  0] = c00[r] + bv0;
                C[base + col + 16] = c01[r] + bv1;
                C[base + col + 32] = c02[r] + bv2;
                C[base + col + 48] = c03[r] + bv3;
            }
            if (rb1 + r < Mstore) {
                const size_t base = (size_t)(rb1 + r) * Ncols;
                C[base + col +  0] = c10[r] + bv0;
                C[base + col + 16] = c11[r] + bv1;
                C[base + col + 32] = c12[r] + bv2;
                C[base + col + 48] = c13[r] + bv3;
            }
        }
    }
}

// ---------------- elementwise / graph kernels ------------------------------
__global__ void k_fill_f32(float* __restrict__ p, float v, size_t n) {
    size_t i = (size_t)blockIdx.x * blockDim.x + threadIdx.x;
    if (i < n) p[i] = v;
}

__global__ void k_fill_f16(_Float16* __restrict__ p, size_t n) {
    size_t i = (size_t)blockIdx.x * blockDim.x + threadIdx.x;
    if (i < n) p[i] = (_Float16)0.0f;
}

__global__ void k_deg_count(float* __restrict__ deg, const int* __restrict__ dst, int e) {
    int i = blockIdx.x * blockDim.x + threadIdx.x;
    if (i < e) atomicAdd(&deg[dst[i]], 1.0f);
}

__global__ void k_rsqrt(const float* __restrict__ deg, float* __restrict__ dinv, int n) {
    int i = blockIdx.x * blockDim.x + threadIdx.x;
    if (i < n) dinv[i] = rsqrtf(deg[i]);
}

__global__ void k_f32_to_f16(const float* __restrict__ in, _Float16* __restrict__ out, size_t n) {
    size_t i = (size_t)blockIdx.x * blockDim.x + threadIdx.x;
    if (i < n) out[i] = (_Float16)in[i];
}

// W[K][N] f32 -> Wt[N][K] f16 (writes coalesced along K)
__global__ void k_transpose_cvt(const float* __restrict__ W, _Float16* __restrict__ Wt,
                                int K, int Ncols) {
    int t = blockIdx.x * blockDim.x + threadIdx.x;
    if (t >= K * Ncols) return;
    int n = t / K, k = t - n * K;
    Wt[t] = (_Float16)W[(size_t)k * Ncols + n];
}

// agg[dst] += dinv[dst] * xw_scaled[src]   (xw already scaled by dinv[src])
__global__ __launch_bounds__(128)
void k_edge_scatter(const float* __restrict__ xw, float* __restrict__ agg,
                    const int* __restrict__ src, const int* __restrict__ dst,
                    const float* __restrict__ dinv, int F4, int F) {
    const int e = blockIdx.x;
    const int s = src[e];
    const int d = dst[e];
    const float c = dinv[d];
    const float4* xr = (const float4*)(xw + (size_t)s * F);
    float*        ar = agg + (size_t)d * F;
    for (int i = threadIdx.x; i < F4; i += blockDim.x) {
        const float4 v = xr[i];
        atomicAdd(ar + 4 * i + 0, c * v.x);
        atomicAdd(ar + 4 * i + 1, c * v.y);
        atomicAdd(ar + 4 * i + 2, c * v.z);
        atomicAdd(ar + 4 * i + 3, c * v.w);
    }
}

// out16 = (relu?)( agg + xw_scaled * dinv[i] + bias )
__global__ __launch_bounds__(256)
void k_combine(const float* __restrict__ xw, const float* __restrict__ agg,
               const float* __restrict__ bias, const float* __restrict__ dinv,
               _Float16* __restrict__ out, int F, int relu) {
    const int i  = blockIdx.x;
    const float di = dinv[i];
    const size_t base = (size_t)i * F;
    for (int f = threadIdx.x; f < F; f += blockDim.x) {
        float v = agg[base + f] + xw[base + f] * di + bias[f];
        if (relu) v = fmaxf(v, 0.0f);
        out[base + f] = (_Float16)v;
    }
}

// ---------------------------------------------------------------------------
static inline size_t ws_take(size_t& off, size_t bytes) {
    off = (off + 255) & ~(size_t)255;
    size_t cur = off;
    off += bytes;
    return cur;
}

extern "C" void kernel_launch(void* const* d_in, const int* in_sizes, int n_in,
                              void* d_out, int out_size, void* d_ws, size_t ws_size,
                              hipStream_t stream) {
    (void)in_sizes; (void)n_in; (void)out_size; (void)ws_size;
    const float* x   = (const float*)d_in[0];
    const int*   ei  = (const int*)  d_in[1];
    const float* W1  = (const float*)d_in[2];
    const float* b1  = (const float*)d_in[3];
    const float* W2  = (const float*)d_in[4];
    const float* b2  = (const float*)d_in[5];
    const float* Wm1 = (const float*)d_in[6];
    const float* bm1 = (const float*)d_in[7];
    const float* Wm2 = (const float*)d_in[8];
    const float* bm2 = (const float*)d_in[9];
    const int* src = ei;
    const int* dst = ei + EE;

    char* ws = (char*)d_ws;
    size_t off = 0;
    float*    deg  = (float*)   (ws + ws_take(off, (size_t)MPAD * 4));
    float*    dinv = (float*)   (ws + ws_take(off, (size_t)MPAD * 4));
    _Float16* x16  = (_Float16*)(ws + ws_take(off, (size_t)MPAD * DD * 2)); // later g16
    _Float16* W1t  = (_Float16*)(ws + ws_take(off, (size_t)DD * HH * 2));
    _Float16* W2t  = (_Float16*)(ws + ws_take(off, (size_t)HH * DD * 2));
    _Float16* Wm1t = (_Float16*)(ws + ws_take(off, (size_t)DD * HH * 2));
    _Float16* Wm2t = (_Float16*)(ws + ws_take(off, (size_t)HH * DD * 2));
    float*    bufA = (float*)   (ws + ws_take(off, (size_t)MPAD * HH * 4)); // GEMM out (row-scaled)
    float*    bufB = (float*)   (ws + ws_take(off, (size_t)MPAD * HH * 4)); // agg; later t16 (f16)
    _Float16* h16  = (_Float16*)(ws + ws_take(off, (size_t)MPAD * HH * 2));
    _Float16* g16  = x16;                    // layer-2 output, f16 (MPAD x D)
    _Float16* t16  = (_Float16*)bufB;        // MLP hidden, f16 (MPAD x H)

    auto cdiv = [](long a, long b) { return (int)((a + b - 1) / b); };
    const int gemm_blk_H = cdiv((long)(MPAD / 32) * (HH / 64), 8);  // waves/8
    const int gemm_blk_D = cdiv((long)(MPAD / 32) * (DD / 64), 8);

    // --- zero the padded tails (keeps pad rows of A inputs at 0) ---
    k_fill_f16<<<cdiv((long)MPAD * DD, 256), 256, 0, stream>>>(x16, (size_t)MPAD * DD);
    k_fill_f16<<<cdiv((long)MPAD * HH, 256), 256, 0, stream>>>(h16, (size_t)MPAD * HH);

    // --- degree / normalization (pad rows: deg=1 -> dinv=1, harmless) ---
    k_fill_f32 <<<cdiv(MPAD, 256), 256, 0, stream>>>(deg, 1.0f, (size_t)MPAD);
    k_deg_count<<<cdiv(EE, 256), 256, 0, stream>>>(deg, dst, EE);
    k_rsqrt    <<<cdiv(MPAD, 256), 256, 0, stream>>>(deg, dinv, MPAD);

    // --- precision conversion (f16 operands for WMMA) ---
    k_f32_to_f16   <<<cdiv((long)NN * DD, 256), 256, 0, stream>>>(x, x16, (size_t)NN * DD);
    k_transpose_cvt<<<cdiv((long)DD * HH, 256), 256, 0, stream>>>(W1,  W1t,  DD, HH);
    k_transpose_cvt<<<cdiv((long)HH * DD, 256), 256, 0, stream>>>(W2,  W2t,  HH, DD);
    k_transpose_cvt<<<cdiv((long)DD * HH, 256), 256, 0, stream>>>(Wm1, Wm1t, DD, HH);
    k_transpose_cvt<<<cdiv((long)HH * DD, 256), 256, 0, stream>>>(Wm2, Wm2t, HH, DD);

    // --- GCN layer 1: xw' = (x @ W1) * dinv[row] ; scatter ; combine + relu ---
    gemm_wmma<EPI_SCALE_ROW_F32, DD><<<gemm_blk_H, 256, 0, stream>>>(x16, W1t, dinv, bufA, MPAD, HH, MPAD);
    k_fill_f32    <<<cdiv((long)NN * HH, 256), 256, 0, stream>>>(bufB, 0.0f, (size_t)NN * HH);
    k_edge_scatter<<<EE, 128, 0, stream>>>(bufA, bufB, src, dst, dinv, HH / 4, HH);
    k_combine     <<<NN, 256, 0, stream>>>(bufA, bufB, b1, dinv, h16, HH, 1);

    // --- GCN layer 2 (no relu) ---
    gemm_wmma<EPI_SCALE_ROW_F32, HH><<<gemm_blk_D, 256, 0, stream>>>(h16, W2t, dinv, bufA, MPAD, DD, MPAD);
    k_fill_f32    <<<cdiv((long)NN * DD, 256), 256, 0, stream>>>(bufB, 0.0f, (size_t)NN * DD);
    k_edge_scatter<<<EE, 128, 0, stream>>>(bufA, bufB, src, dst, dinv, DD / 4, DD);
    k_combine     <<<NN, 256, 0, stream>>>(bufA, bufB, b2, dinv, g16, DD, 0);

    // --- MLP: out = relu(g @ Wm1 + bm1) @ Wm2 + bm2 ---
    gemm_wmma<EPI_BIAS_RELU_F16, DD><<<gemm_blk_H, 256, 0, stream>>>(g16, Wm1t, bm1, t16, MPAD, HH, MPAD);
    gemm_wmma<EPI_BIAS_F32,      HH><<<gemm_blk_D, 256, 0, stream>>>(t16, Wm2t, bm2, d_out, MPAD, DD, NN);
}
